// CRF_73830487818259
// MI455X (gfx1250) — compile-verified, matrix operand order checked by reading, not run on hardware
//
#include <hip/hip_runtime.h>

// ---------------- problem constants ----------------
#define E     512
#define HID   512
#define Hh    256          // per-direction hidden
#define FOURH 1024         // 4*Hh gate rows
#define SLEN  8192
#define TAGS  16
#define START_ID 14
#define STOP_ID  15
#define NEGV  (-10000.0f)

typedef _Float16 v16h __attribute__((ext_vector_type(16)));
typedef _Float16 v8h  __attribute__((ext_vector_type(8)));
typedef float    v8f  __attribute__((ext_vector_type(8)));

__device__ __forceinline__ float fast_sig(float x)  { return 1.0f / (1.0f + __expf(-x)); }
__device__ __forceinline__ float fast_tanh(float x) { return 1.0f - 2.0f / (1.0f + __expf(2.0f * x)); }

// Assemble a 16-element f16 WMMA fragment from two aligned 8-element runs.
__device__ __forceinline__ v16h frag16(const _Float16* p, int k0) {
  const v8h lo = *(const v8h*)(p + k0);
  const v8h hi = *(const v8h*)(p + k0 + 16);
  return __builtin_shufflevector(lo, hi, 0,1,2,3,4,5,6,7,8,9,10,11,12,13,14,15);
}

// =====================================================================
// Prep A: gather sentence embeddings to f16 [SLEN][E] (forward order).
// =====================================================================
__global__ __launch_bounds__(256) void embed_gather_kernel(
    const int* __restrict__ sentence, const float* __restrict__ embedding,
    _Float16* __restrict__ embeds)
{
  const int s = blockIdx.x;
  const float* src = embedding + (long)sentence[s] * E;
  _Float16* dst = embeds + (long)s * E;
  for (int k = threadIdx.x; k < E; k += 256) dst[k] = (_Float16)src[k];
}

// Prep B: generic f32 -> f16 conversion (weights).
__global__ __launch_bounds__(256) void cvt_f16_kernel(
    const float* __restrict__ src, _Float16* __restrict__ dst, int n)
{
  for (int i = blockIdx.x * 256 + threadIdx.x; i < n; i += gridDim.x * 256)
    dst[i] = (_Float16)src[i];
}

// =====================================================================
// Kernel 1: xg[dir][s][j] = embeds[src(s)] @ Wih_dir^T + (bih+bhh)
// Block = 256 thr (8 waves). Each wave: one 16(M) x 64(N) strip —
// 4 N-tiles sharing the A fragment, 4 independent WMMA accumulators.
// Grid: x = 4H/512 = 2, y = S/16 = 512, z = dir.
// =====================================================================
__global__ __launch_bounds__(256) void xgemm_kernel(
    const _Float16* __restrict__ embeds,   // [SLEN][E] f16 (forward order)
    const _Float16* __restrict__ Wf16,     // [2][FOURH][E] f16
    const float* __restrict__ bih_f, const float* __restrict__ bhh_f,
    const float* __restrict__ bih_b, const float* __restrict__ bhh_b,
    float*       __restrict__ xg)          // [2][SLEN][FOURH]
{
  const int dir  = blockIdx.z;
  const int wave = threadIdx.x >> 5;
  const int lane = threadIdx.x & 31;
  const int half = lane >> 4;        // K window +0 / +8
  const int lrow = lane & 15;
  const int m0 = blockIdx.y * 16;
  const int nb = blockIdx.x * 512 + wave * 64;   // wave's 64-col strip

  const _Float16* Wd  = Wf16 + (long)dir * FOURH * E;
  const float*    bih = dir ? bih_b : bih_f;
  const float*    bhh = dir ? bhh_b : bhh_f;

  // A row: time index, reversed for the backward direction
  const int t_row = m0 + lrow;
  const _Float16* arow = embeds + (long)(dir ? (SLEN - 1 - t_row) : t_row) * E;
  const _Float16* brow[4];
#pragma unroll
  for (int nt = 0; nt < 4; ++nt)
    brow[nt] = Wd + (long)(nb + nt * 16 + lrow) * E;

  v8f acc[4] = {};
  for (int kb = 0; kb < E; kb += 32) {
    const int k0 = kb + (half ? 8 : 0);
    const v16h a = frag16(arow, k0);
#pragma unroll
    for (int nt = 0; nt < 4; ++nt) {
      const v16h b = frag16(brow[nt], k0);
      acc[nt] = __builtin_amdgcn_wmma_f32_16x16x32_f16(false, a, false, b,
                                                       (short)0, acc[nt], false, false);
    }
  }

  float* outp = xg + (long)dir * SLEN * FOURH;
#pragma unroll
  for (int nt = 0; nt < 4; ++nt) {
    const int   col  = nb + nt * 16 + lrow;
    const float bias = bih[col] + bhh[col];
#pragma unroll
    for (int r = 0; r < 8; ++r) {
      const int row = m0 + r + (half ? 8 : 0);
      outp[(long)row * FOURH + col] = acc[nt][r] + bias;
    }
  }
}

// =====================================================================
// Kernel 2: persistent bidirectional LSTM recurrence.
// 2 blocks (one per direction) x 1024 threads (fills a WGP).
// Thread j owns gate row j of Whh as 128 packed-f16 dwords in VGPRs.
// h broadcast via LDS; emits hs in f16 for the downstream WMMA.
// =====================================================================
__global__ __launch_bounds__(1024) void lstm_kernel(
    const float* __restrict__ Whh_f, const float* __restrict__ Whh_b,
    const float* __restrict__ h0,    const float* __restrict__ c0,
    const float* __restrict__ xg,    // [2][SLEN][FOURH]
    _Float16*    __restrict__ hs)    // [SLEN][HID] f16; fwd cols 0..255, bwd 256..511
{
  const int dir = blockIdx.x;
  const int j   = threadIdx.x;                 // gate row 0..1023
  const float* Whh = dir ? Whh_b : Whh_f;
  const float* xgd = xg + (long)dir * SLEN * FOURH;

  __shared__ float h_lds[Hh];
  __shared__ float g_lds[FOURH];

  // Weight-stationary: pack this thread's Whh row as f16 pairs (128 VGPRs).
  unsigned wreg[Hh / 2];
  const float2* wrow = (const float2*)(Whh + (long)j * Hh);
#pragma unroll 16
  for (int k2 = 0; k2 < Hh / 2; ++k2) {
    float2 w = wrow[k2];
    unsigned short u0 = __builtin_bit_cast(unsigned short, (_Float16)w.x);
    unsigned short u1 = __builtin_bit_cast(unsigned short, (_Float16)w.y);
    wreg[k2] = (unsigned)u0 | ((unsigned)u1 << 16);
  }

  float c = 0.0f;
  if (j < Hh) {
    h_lds[j] = h0[dir * Hh + j];
    c        = c0[dir * Hh + j];
  }
  __syncthreads();

  const float2* h2 = (const float2*)h_lds;
  for (int t = 0; t < SLEN; ++t) {
    float acc = xgd[(long)t * FOURH + j];
    if (t + 1 < SLEN) __builtin_prefetch(&xgd[(long)(t + 1) * FOURH + j], 0, 3);
#pragma unroll 16
    for (int k2 = 0; k2 < Hh / 2; ++k2) {
      const unsigned u = wreg[k2];
      const _Float16 w0 = __builtin_bit_cast(_Float16, (unsigned short)(u & 0xffffu));
      const _Float16 w1 = __builtin_bit_cast(_Float16, (unsigned short)(u >> 16));
      const float2 hv = h2[k2];
      acc += (float)w0 * hv.x + (float)w1 * hv.y;
    }
    g_lds[j] = acc;
    __syncthreads();
    if (j < Hh) {
      const float ig = fast_sig (g_lds[j]);
      const float fg = fast_sig (g_lds[Hh + j]);
      const float gg = fast_tanh(g_lds[2 * Hh + j]);
      const float og = fast_sig (g_lds[3 * Hh + j]);
      c = fg * c + ig * gg;
      const float hval = og * fast_tanh(c);
      h_lds[j] = hval;
      const int row = dir ? (SLEN - 1 - t) : t;
      hs[(long)row * HID + dir * Hh + j] = (_Float16)hval;
    }
    __syncthreads();
  }
}

// =====================================================================
// Kernel 3: feats = hs @ W_tag^T + b_tag   ([8192,512] x [512,16])
// N = 16 = one WMMA tile wide; dual accumulators (even/odd K chunks)
// to pipeline the WMMA chain. Block 256 thr = 8 waves; grid = 64.
// =====================================================================
__global__ __launch_bounds__(256) void feats_kernel(
    const _Float16* __restrict__ hs, const _Float16* __restrict__ Wtag16,
    const float* __restrict__ b_tag, float* __restrict__ feats)
{
  const int wave = threadIdx.x >> 5;
  const int lane = threadIdx.x & 31;
  const int half = lane >> 4;
  const int lrow = lane & 15;
  const int m0 = (blockIdx.x * 8 + wave) * 16;

  const _Float16* arow = hs     + (long)(m0 + lrow) * HID;
  const _Float16* brow = Wtag16 + (long)lrow * HID;    // N (tag) = lrow

  v8f acc0 = {}, acc1 = {};
  for (int kb = 0; kb < HID; kb += 64) {
    const int k0 = kb + (half ? 8 : 0);
    acc0 = __builtin_amdgcn_wmma_f32_16x16x32_f16(
        false, frag16(arow, k0),      false, frag16(brow, k0),
        (short)0, acc0, false, false);
    acc1 = __builtin_amdgcn_wmma_f32_16x16x32_f16(
        false, frag16(arow, k0 + 32), false, frag16(brow, k0 + 32),
        (short)0, acc1, false, false);
  }

  const float bias = b_tag[lrow];
#pragma unroll
  for (int r = 0; r < 8; ++r) {
    const int row = m0 + r + (half ? 8 : 0);
    feats[row * TAGS + lrow] = acc0[r] + acc1[r] + bias;
  }
}

// =====================================================================
// Kernel 4: Viterbi max-plus scan + backtrack. One wave (32 lanes).
// =====================================================================
__global__ __launch_bounds__(32) void viterbi_kernel(
    const float* __restrict__ feats, const float* __restrict__ transitions,
    int* __restrict__ bp, float* __restrict__ out)
{
  const int lane = threadIdx.x;
  const int n    = lane & 15;

  float trn[TAGS];
#pragma unroll
  for (int p = 0; p < TAGS; ++p) trn[p] = transitions[n * TAGS + p];

  float fv = (n == START_ID) ? 0.0f : NEGV;

  for (int t = 0; t < SLEN; ++t) {
    float best = NEGV * 4.0f;
    int bestp = 0;
#pragma unroll
    for (int p = 0; p < TAGS; ++p) {
      const float cand = __shfl(fv, p, 32) + trn[p];
      if (cand > best) { best = cand; bestp = p; }
    }
    if (lane < 16) bp[t * TAGS + n] = bestp;
    fv = best + feats[t * TAGS + n];
  }

  float val = (lane < 16) ? (fv + transitions[STOP_ID * TAGS + n]) : NEGV * 4.0f;
  int   idx = n;
#pragma unroll
  for (int off = 16; off > 0; off >>= 1) {
    const float ov = __shfl_xor(val, off, 32);
    const int   oi = __shfl_xor(idx, off, 32);
    if (ov > val) { val = ov; idx = oi; }
  }

  __threadfence();   // our bp stores -> our backtrack loads

  if (lane == 0) {
    out[0] = val;                          // path_score
    int tag = idx;                         // last_tag
    for (int t = SLEN - 1; t >= 0; --t) {  // emit tag, then follow backpointer
      out[1 + t] = (float)tag;
      tag = bp[t * TAGS + tag];
    }
  }
}

// =====================================================================
extern "C" void kernel_launch(void* const* d_in, const int* in_sizes, int n_in,
                              void* d_out, int out_size, void* d_ws, size_t ws_size,
                              hipStream_t stream) {
  (void)in_sizes; (void)n_in; (void)out_size; (void)ws_size;
  const int*   sentence    = (const int*)  d_in[0];
  const float* embedding   = (const float*)d_in[1];
  const float* Wih_f       = (const float*)d_in[2];
  const float* Whh_f       = (const float*)d_in[3];
  const float* bih_f       = (const float*)d_in[4];
  const float* bhh_f       = (const float*)d_in[5];
  const float* Wih_b       = (const float*)d_in[6];
  const float* Whh_b       = (const float*)d_in[7];
  const float* bih_b       = (const float*)d_in[8];
  const float* bhh_b       = (const float*)d_in[9];
  const float* h0          = (const float*)d_in[10];
  const float* c0          = (const float*)d_in[11];
  const float* W_tag       = (const float*)d_in[12];
  const float* b_tag       = (const float*)d_in[13];
  const float* transitions = (const float*)d_in[14];
  float* out = (float*)d_out;

  // ---- workspace layout (16B-aligned regions) ----
  char* ws = (char*)d_ws;
  float*    xg     = (float*)ws;              ws += 2L * SLEN * FOURH * 4;  // 64 MB
  float*    feats  = (float*)ws;              ws += (long)SLEN * TAGS * 4;
  int*      bp     = (int*)ws;                ws += (long)SLEN * TAGS * 4;
  _Float16* embeds = (_Float16*)ws;           ws += (long)SLEN * E * 2;     // 8 MB
  _Float16* Wf16   = (_Float16*)ws;           ws += 2L * FOURH * E * 2;     // 2 MB
  _Float16* Wtag16 = (_Float16*)ws;           ws += (long)TAGS * HID * 2;
  _Float16* hs     = (_Float16*)ws;           ws += (long)SLEN * HID * 2;   // 8 MB

  // 0) stage f16 operands (gathered embeds + weights)
  embed_gather_kernel<<<SLEN, 256, 0, stream>>>(sentence, embedding, embeds);
  cvt_f16_kernel<<<512, 256, 0, stream>>>(Wih_f, Wf16,              FOURH * E);
  cvt_f16_kernel<<<512, 256, 0, stream>>>(Wih_b, Wf16 + FOURH * E,  FOURH * E);
  cvt_f16_kernel<<<32,  256, 0, stream>>>(W_tag, Wtag16,            TAGS * HID);

  // 1) input projections, both directions (WMMA, 4 tiles/wave)
  xgemm_kernel<<<dim3(FOURH / 512, SLEN / 16, 2), 256, 0, stream>>>(
      embeds, Wf16, bih_f, bhh_f, bih_b, bhh_b, xg);

  // 2) persistent recurrence, one WGP-filling block per direction
  lstm_kernel<<<2, 1024, 0, stream>>>(Whh_f, Whh_b, h0, c0, xg, hs);

  // 3) tag projection (WMMA, N = 16 tile, dual accumulators)
  feats_kernel<<<SLEN / 128, 256, 0, stream>>>(hs, Wtag16, b_tag, feats);

  // 4) Viterbi scan + backtrack (single wave)
  viterbi_kernel<<<1, 32, 0, stream>>>(feats, transitions, bp, out);
}
